// JointNetwork_24309514896038
// MI455X (gfx1250) — matrix operand order repaired and checked by reading, not compiled, not run on hardware
//
#include <hip/hip_runtime.h>
#include <hip/hip_bf16.h>

// RNN-T joint network: out[b,t,u,v] = tanh(enc[b,t,:] + dec[b,u,:]) . W_out + b_out
// B=4 T=256 U=64, D=J=V=512.  All heavy math on v_wmma_f32_16x16x32_f16.

typedef __attribute__((ext_vector_type(16))) _Float16 v16h;
typedef __attribute__((ext_vector_type(8)))  _Float16 v8h;
typedef __attribute__((ext_vector_type(8)))  float    v8f;
typedef __attribute__((ext_vector_type(4)))  float    v4f;
typedef __attribute__((ext_vector_type(4)))  int      v4i;

#define KD 512   // inner dim (= D_ENC = D_DEC = J)
#define ND 512   // output dim (= J for enc/dec GEMMs, = V for joint GEMM)

#if __has_builtin(__builtin_amdgcn_global_load_async_to_lds_b128) && \
    __has_builtin(__builtin_amdgcn_s_wait_asynccnt)
#define USE_ASYNC_LDS 1
// Parameter types per hipcc diagnostic: v4i pointers, param0 in AS1 (global).
typedef __attribute__((address_space(1))) v4i gv4i;
typedef __attribute__((address_space(3))) v4i lv4i;
#else
#define USE_ASYNC_LDS 0
#endif

// Branch-free tanh: feeds an f16 WMMA operand, so v_exp_f32/v_rcp_f32 accuracy
// (~1 ulp) is far beyond what f16 can represent.  ~7 VALU ops, no SALU.
__device__ __forceinline__ float fast_tanh(float x) {
    float xc = fminf(fmaxf(x, -10.0f), 10.0f);            // avoid inf*0 -> NaN
    float y  = __builtin_amdgcn_exp2f(xc * 2.885390081777927f);  // e^(2x)
    return (y - 1.0f) * __builtin_amdgcn_rcpf(y + 1.0f);
}

// ---------------------------------------------------------------------------
// Pack a 512x512 f32 row-major weight matrix (k-major) into f16 B-fragments
// for V_WMMA_F32_16X16X32_F16.  B (32x16, KxN) per ISA 7.12.2:
//   lane l holds column n = l&15;  halves = k {kg*8..+7} U {16+kg*8..+7}, kg=l>>4.
// Flat layout: frag(kblk, nblk, lane)[16 halves], 32B per lane entry.
// ---------------------------------------------------------------------------
__global__ __launch_bounds__(256) void pack_b_kernel(const float* __restrict__ W,
                                                     _Float16* __restrict__ dst) {
    const int idx  = blockIdx.x * blockDim.x + threadIdx.x;   // 16*32*32 = 16384
    const int lane = idx & 31;
    const int nblk = (idx >> 5) & 31;
    const int kblk = idx >> 10;
    const int n     = nblk * 16 + (lane & 15);
    const int kbase = kblk * 32 + (lane >> 4) * 8;
    v16h frag;
#pragma unroll
    for (int h = 0; h < 16; ++h)
        frag[h] = (_Float16)W[(size_t)(kbase + h + (h & 8)) * ND + n];
    *(v16h*)(dst + (size_t)idx * 16) = frag;                  // 32B aligned
}

// ---------------------------------------------------------------------------
// Dense GEMM: out(M x 512) = A(M x 512, f32) * Bpk(+ bias).  One 16-row tile
// per wave, 8 waves per block.  A-fragment per ISA 7.12.2 (16-bit A 16x32):
//   lane l holds row m = l&15;  halves = k {kg*8..+7} U {16+kg*8..+7}, kg=l>>4.
// ---------------------------------------------------------------------------
template<bool HAS_BIAS>
__global__ __launch_bounds__(256) void gemm16_wmma(const float* __restrict__ A,
                                                   const _Float16* __restrict__ Bpk,
                                                   const float* __restrict__ bias,
                                                   float* __restrict__ out) {
    const int lane = threadIdx.x & 31;
    const int w    = threadIdx.x >> 5;
    const int kg   = lane >> 4;
    const int ml   = lane & 15;
    const int mt   = blockIdx.x * 8 + w;        // 16-row tile id
    const float* arow = A + ((size_t)mt * 16 + ml) * KD;

    v16h af[16];
#pragma unroll
    for (int k0 = 0; k0 < 16; ++k0) {
        const float* p = arow + k0 * 32 + kg * 8;
        v4f a0 = *(const v4f*)(p);
        v4f a1 = *(const v4f*)(p + 4);
        v4f a2 = *(const v4f*)(p + 16);
        v4f a3 = *(const v4f*)(p + 20);
#pragma unroll
        for (int h = 0; h < 4; ++h) {
            af[k0][h]      = (_Float16)a0[h];
            af[k0][h + 4]  = (_Float16)a1[h];
            af[k0][h + 8]  = (_Float16)a2[h];
            af[k0][h + 12] = (_Float16)a3[h];
        }
    }

    for (int nblk = 0; nblk < 32; ++nblk) {
        v8f acc = {};
#pragma unroll
        for (int k0 = 0; k0 < 16; ++k0) {
            v16h bf = *(const v16h*)(Bpk + ((size_t)(k0 * 32 + nblk) * 32 + lane) * 16);
            acc = __builtin_amdgcn_wmma_f32_16x16x32_f16(false, af[k0], false, bf,
                                                         (short)0, acc, false, false);
        }
        const int n  = nblk * 16 + ml;
        const float bv = HAS_BIAS ? bias[n] : 0.0f;
        const size_t rowbase = (size_t)mt * 16 + kg * 8;   // C/D: m=(l>>4)*8+r, n=l&15
#pragma unroll
        for (int r = 0; r < 8; ++r)
            out[(rowbase + r) * ND + n] = acc[r] + bv;
    }
}

// ---------------------------------------------------------------------------
// Joint kernel.  Wave mt (0..4095) owns rows [mt*16, mt*16+16) of the flat
// (b,t,u) M dimension: bt = mt>>2, u-subtile = mt&3.  A = tanh(enc+dec) built
// once in registers (each z element computed exactly once globally).  W_out
// fragments double-buffered through LDS: each block reads W_out once (8x L2
// traffic cut).  24-half lane stride in LDS avoids ds bank conflicts.
// Staging uses GLOBAL_LOAD_ASYNC_TO_LDS_B128 when available (ASYNCcnt path,
// bypasses VGPRs), else plain load+ds_store.
// ---------------------------------------------------------------------------
__global__ __launch_bounds__(256) void joint_wmma(const float* __restrict__ enc,
                                                  const float* __restrict__ dec,
                                                  const _Float16* __restrict__ Bpk,
                                                  const float* __restrict__ bias,
                                                  float* __restrict__ out) {
    const int tid  = threadIdx.x;
    const int lane = tid & 31;
    const int w    = tid >> 5;
    const int kg   = lane >> 4;
    const int ml   = lane & 15;
    const int mt   = blockIdx.x * 8 + w;   // 16-row tile of M = B*T*U
    const int bt   = mt >> 2;              // (b,t) pair 0..1023
    const int b    = bt >> 8;
    const int u    = (mt & 3) * 16 + ml;

    const float* erow = enc + (size_t)bt * KD;
    const float* drow = dec + ((size_t)b * 64 + u) * KD;

    // Build A-fragments: z = tanh(enc + dec), f32 -> f16
    v16h af[16];
#pragma unroll
    for (int k0 = 0; k0 < 16; ++k0) {
        const int kb = k0 * 32 + kg * 8;
        v4f e0 = *(const v4f*)(erow + kb);
        v4f e1 = *(const v4f*)(erow + kb + 4);
        v4f e2 = *(const v4f*)(erow + kb + 16);
        v4f e3 = *(const v4f*)(erow + kb + 20);
        v4f d0 = *(const v4f*)(drow + kb);
        v4f d1 = *(const v4f*)(drow + kb + 4);
        v4f d2 = *(const v4f*)(drow + kb + 16);
        v4f d3 = *(const v4f*)(drow + kb + 20);
#pragma unroll
        for (int h = 0; h < 4; ++h) {
            af[k0][h]      = (_Float16)fast_tanh(e0[h] + d0[h]);
            af[k0][h + 4]  = (_Float16)fast_tanh(e1[h] + d1[h]);
            af[k0][h + 8]  = (_Float16)fast_tanh(e2[h] + d2[h]);
            af[k0][h + 12] = (_Float16)fast_tanh(e3[h] + d3[h]);
        }
    }

    // LDS B staging: [buf][k0][lane][24 halves] (16 used, 24-stride = no conflicts)
    __shared__ __align__(16) _Float16 bsh[2 * 16 * 32 * 24];

    auto stage = [&](int nblk, int buf) {
#pragma unroll
        for (int i = 0; i < 2; ++i) {
            const int e  = tid * 2 + i;        // 512 lane-entries per slab
            const int k0 = e >> 5, ln = e & 31;
            const _Float16* src = Bpk + ((size_t)(k0 * 32 + nblk) * 32 + ln) * 16;
            _Float16* dst = &bsh[(size_t)((buf * 16 + k0) * 32 + ln) * 24];
#if USE_ASYNC_LDS
            __builtin_amdgcn_global_load_async_to_lds_b128(
                (gv4i*)(src), (lv4i*)(dst), 0, 0);
            __builtin_amdgcn_global_load_async_to_lds_b128(
                (gv4i*)(src + 8), (lv4i*)(dst + 8), 0, 0);
#else
            *(v8h*)dst       = *(const v8h*)src;
            *(v8h*)(dst + 8) = *(const v8h*)(src + 8);
#endif
        }
    };

    stage(0, 0);
#if USE_ASYNC_LDS
    __builtin_amdgcn_s_wait_asynccnt(0);
#endif
    __syncthreads();

    for (int nblk = 0; nblk < 32; ++nblk) {
        const int cur = nblk & 1;
        if (nblk + 1 < 32) stage(nblk + 1, cur ^ 1);   // prefetch overlaps wmma

        v8f acc = {};
#pragma unroll
        for (int k0 = 0; k0 < 16; ++k0) {
            const _Float16* p = &bsh[(size_t)((cur * 16 + k0) * 32 + lane) * 24];
            v8h lo = *(const v8h*)p;
            v8h hi = *(const v8h*)(p + 8);
            v16h bf = __builtin_shufflevector(lo, hi, 0,1,2,3,4,5,6,7,
                                              8,9,10,11,12,13,14,15);
            acc = __builtin_amdgcn_wmma_f32_16x16x32_f16(false, af[k0], false, bf,
                                                         (short)0, acc, false, false);
        }
        const int n  = nblk * 16 + ml;
        const float bv = bias[n];
        const size_t rowbase = (size_t)mt * 16 + kg * 8;
#pragma unroll
        for (int r = 0; r < 8; ++r)
            out[(rowbase + r) * ND + n] = acc[r] + bv;

#if USE_ASYNC_LDS
        __builtin_amdgcn_s_wait_asynccnt(0);   // staged slab fully in LDS
#endif
        __syncthreads();                       // protect double buffer swap
    }
}

// ---------------------------------------------------------------------------
extern "C" void kernel_launch(void* const* d_in, const int* in_sizes, int n_in,
                              void* d_out, int out_size, void* d_ws, size_t ws_size,
                              hipStream_t stream) {
    const float* h_enc = (const float*)d_in[0];  // (1024, 512)
    const float* h_dec = (const float*)d_in[1];  // (256, 512)
    const float* W_enc = (const float*)d_in[2];  // (512, 512)
    const float* b_enc = (const float*)d_in[3];  // (512,)
    const float* W_dec = (const float*)d_in[4];  // (512, 512)
    const float* W_out = (const float*)d_in[5];  // (512, 512)
    const float* b_out = (const float*)d_in[6];  // (512,)
    float* out = (float*)d_out;                  // (4,256,64,512) f32

    char* ws = (char*)d_ws;
    float*    enc_ws = (float*)(ws);                                  // 2 MB
    float*    dec_ws = (float*)(ws + (2u << 20));                     // 0.5 MB
    _Float16* wencp  = (_Float16*)(ws + (2u << 20) + (512u << 10));   // 0.5 MB
    _Float16* wdecp  = wencp + 16 * 32 * 32 * 16;                     // 0.5 MB
    _Float16* woutp  = wdecp + 16 * 32 * 32 * 16;                     // 0.5 MB

    // Pack weights into WMMA B-fragment layout (f32 -> f16)
    pack_b_kernel<<<64, 256, 0, stream>>>(W_enc, wencp);
    pack_b_kernel<<<64, 256, 0, stream>>>(W_dec, wdecp);
    pack_b_kernel<<<64, 256, 0, stream>>>(W_out, woutp);

    // enc = h_enc @ W_enc + b_enc   (1024 rows -> 64 wave-tiles -> 8 blocks)
    gemm16_wmma<true ><<<8, 256, 0, stream>>>(h_enc, wencp, b_enc, enc_ws);
    // dec = h_dec @ W_dec           (256 rows -> 16 wave-tiles -> 2 blocks)
    gemm16_wmma<false><<<2, 256, 0, stream>>>(h_dec, wdecp, nullptr, dec_ws);

    // out = tanh(enc + dec) @ W_out + b_out   (4096 wave-tiles -> 512 blocks)
    joint_wmma<<<512, 256, 0, stream>>>(enc_ws, dec_ws, woutp, b_out, out);
}
